// CharTransformerEmbeddings_53695681134662
// MI455X (gfx1250) — matrix operand order repaired and checked by reading, not compiled, not run on hardware
//
#include <hip/hip_runtime.h>
#include <hip/hip_bf16.h>

// ---------------------------------------------------------------------------
// CDNA5 (gfx1250) char-transformer forward.
// GEMMs: v_wmma_f32_16x16x32_bf16 (bf16 in, f32 accumulate).
// B-tile staging: global_load_async_to_lds_b128 (ASYNCcnt-tracked, no VGPRs).
// ---------------------------------------------------------------------------

typedef __attribute__((ext_vector_type(16))) __bf16 bf16x16;
typedef __attribute__((ext_vector_type(8)))  __bf16 bf16x8;
typedef __attribute__((ext_vector_type(4)))  __bf16 bf16x4;
typedef __attribute__((ext_vector_type(8)))  float  f32x8;
typedef __attribute__((ext_vector_type(4)))  float  f32x4;

#define D_MODEL 256
#define N_HEADS 8
#define DK      32
#define WLEN    16
#define NWORDS  2048           // B*T = 16*128
#define MROWS   32768          // NWORDS * WLEN
#define DFF     1024
#define NLAYERS 2
#define RPR_N   17             // 2K+1
#define NEGINF  (-1e9f)

// A-fragment K index for the 16-bit 16x32 A layout (wave32):
// lanes 0-15 hold K = {0..7, 16..23}, lanes 16-31 hold K = {8..15, 24..31},
// packed 2 bf16 per VGPR.
__device__ __forceinline__ int kmapA(int e, int half) {
    int g = e >> 1, p = e & 1;
    return (g < 4 ? 2 * g : 16 + 2 * (g - 4)) + p + 8 * half;
}

// Async 16-byte global -> LDS copy (per lane). Tracked by ASYNCcnt.
// LDS operand is the wave-relative LDS byte offset (= low 32 bits of the
// generic pointer for flat-LDS addressing, ISA §10.2).
__device__ __forceinline__ void async_copy_b128(void* lds_dst, const void* gsrc) {
    unsigned lds_off = (unsigned)(unsigned long long)lds_dst;
    unsigned long long gaddr = (unsigned long long)gsrc;
    asm volatile("global_load_async_to_lds_b128 %0, %1, off"
                 :
                 : "v"(lds_off), "v"(gaddr)
                 : "memory");
}

__device__ __forceinline__ void wait_asynccnt0() {
    asm volatile("s_wait_asynccnt 0x0" ::: "memory");
}

// ---------------------------------------------------------------------------
// Elementwise helpers
// ---------------------------------------------------------------------------
__global__ void cvt_bf16_kernel(const float* __restrict__ src,
                                __bf16* __restrict__ dst, int n) {
    int i = blockIdx.x * blockDim.x + threadIdx.x;
    if (i < n) dst[i] = (__bf16)src[i];
}

__global__ void lengths_kernel(const int* __restrict__ xch,
                               int* __restrict__ len, int nwords) {
    int n = blockIdx.x * blockDim.x + threadIdx.x;
    if (n < nwords) {
        int c = 0;
#pragma unroll
        for (int w = 0; w < WLEN; ++w) c += (xch[n * WLEN + w] != 0) ? 1 : 0;
        len[n] = c;
    }
}

// one block per char-position row, one thread per model dim
__global__ void embed_kernel(const int* __restrict__ xch,
                             const float* __restrict__ emb,
                             float* __restrict__ out) {
    int row = blockIdx.x;
    int d = threadIdx.x;
    int ch = xch[row];
    out[row * D_MODEL + d] = emb[ch * D_MODEL + d];
}

// LayerNorm: one block (256 threads) per row of 256 elements
__global__ void ln_kernel(const float* __restrict__ X,
                          const float* __restrict__ g,
                          const float* __restrict__ b,
                          float* __restrict__ Y) {
    __shared__ float s1[256];
    __shared__ float s2[256];
    int row = blockIdx.x, t = threadIdx.x;
    float v = X[row * D_MODEL + t];
    s1[t] = v;
    s2[t] = v * v;
    __syncthreads();
    for (int off = 128; off > 0; off >>= 1) {
        if (t < off) { s1[t] += s1[t + off]; s2[t] += s2[t + off]; }
        __syncthreads();
    }
    float m   = s1[0] * (1.0f / D_MODEL);
    float var = s2[0] * (1.0f / D_MODEL) - m * m;
    float r   = rsqrtf(var + 1e-5f);
    Y[row * D_MODEL + t] = (v - m) * r * g[t] + b[t];
}

// max-pool over the W=16 char positions of each word
__global__ void maxpool_kernel(const float* __restrict__ X,
                               float* __restrict__ out) {
    int n = blockIdx.x, d = threadIdx.x;
    float m = X[(n * WLEN) * D_MODEL + d];
#pragma unroll
    for (int w = 1; w < WLEN; ++w)
        m = fmaxf(m, X[(n * WLEN + w) * D_MODEL + d]);
    out[n * D_MODEL + d] = m;
}

// ---------------------------------------------------------------------------
// Tiled bf16 WMMA GEMM:  Y[M,N] = epilogue( X[M,K] @ Wb[K,N] + bias )
// RELU / RES are compile-time so the epilogue is branch-free.
// Block: 256 threads = 8 waves. BM=128 (16 rows/wave), BN=64, BK=32.
// A staged via float4 loads + f32->bf16 pack; B staged via async b128 -> LDS.
// ---------------------------------------------------------------------------
#define BM 128
#define BN 64
#define BK 32
#define AS_STRIDE 36   // multiple of 4 -> 8B-aligned bf16x4 stores, padded
#define BS_STRIDE 72   // multiple of 8 -> 16B-aligned b128 stores, padded

template <int RELU, int RES>
__global__ void gemm_bf16_wmma(const float* __restrict__ X,
                               const __bf16* __restrict__ Wb,
                               const float* __restrict__ bias,
                               const float* __restrict__ Rres,
                               float* __restrict__ Y,
                               int M, int Kd, int Nd) {
    __shared__ __bf16 As[BM * AS_STRIDE];
    __shared__ __bf16 Bs[BK * BS_STRIDE];

    int tid  = threadIdx.x;
    int wave = tid >> 5;
    int lane = tid & 31;
    int half = lane >> 4;
    int lj   = lane & 15;
    int klane = lj + 16 * half;            // unused; B K index is kmapA(e, half)

    int rowBase = blockIdx.y * BM;
    int colBase = blockIdx.x * BN;

    // B staging: 32x64 = 256 chunks of 8 bf16 (16B), 1 per thread
    int br = tid >> 3;                     // 0..31
    int bc = (tid & 7) * 8;                // 0..56

    f32x8 acc[4] = {};

    for (int k0 = 0; k0 < Kd; k0 += BK) {
        // stage B tile (32x64) bf16 via async global->LDS b128 (no VGPRs)
        async_copy_b128(Bs + br * BS_STRIDE + bc,
                        Wb + (size_t)(k0 + br) * Nd + colBase + bc);

        // stage A tile (128x32), f32 -> bf16, vectorized b128 loads
#pragma unroll
        for (int i = 0; i < 4; ++i) {
            int c  = i * 256 + tid;
            int r  = c >> 3;
            int c4 = (c & 7) * 4;
            const f32x4* src =
                (const f32x4*)(X + (size_t)(rowBase + r) * Kd + k0 + c4);
            f32x4 v4 = *src;
            bf16x4 b4;
            b4.x = (__bf16)v4.x; b4.y = (__bf16)v4.y;
            b4.z = (__bf16)v4.z; b4.w = (__bf16)v4.w;
            *(bf16x4*)(As + r * AS_STRIDE + c4) = b4;
            // prefetch next K-step of A into cache (global_prefetch_b8)
            if (k0 + BK < Kd)
                __builtin_prefetch(X + (size_t)(rowBase + r) * Kd + k0 + BK + c4, 0, 1);
        }

        wait_asynccnt0();     // B tile resident in LDS
        __syncthreads();

        bf16x16 afrag;
#pragma unroll
        for (int e = 0; e < 16; ++e)
            afrag[e] = As[(wave * 16 + lj) * AS_STRIDE + kmapA(e, half)];

#pragma unroll
        for (int t = 0; t < 4; ++t) {
            bf16x16 bfrag;
#pragma unroll
            for (int e = 0; e < 16; ++e)
                bfrag[e] = Bs[kmapA(e, half) * BS_STRIDE + t * 16 + lj];
            acc[t] = __builtin_amdgcn_wmma_f32_16x16x32_bf16(
                false, afrag, false, bfrag, (short)0, acc[t], false, false);
        }
        __syncthreads();
    }

    // branch-free epilogue, per C-layout: m = r + 8*(lane>>4), n = lane&15
#pragma unroll
    for (int t = 0; t < 4; ++t) {
        float bcol = bias[colBase + t * 16 + lj];
#pragma unroll
        for (int r = 0; r < 8; ++r) {
            int m   = r + 8 * half;
            int row = rowBase + wave * 16 + m;
            int col = colBase + t * 16 + lj;
            float v = acc[t][r] + bcol;
            if (RELU) v = fmaxf(v, 0.0f);
            if (RES)  v += Rres[(size_t)row * Nd + col];
            Y[(size_t)row * Nd + col] = v;
        }
    }
}

// ---------------------------------------------------------------------------
// Fused relative-position attention. One block per word (n), one wave per head.
// scores = q@k^T (WMMA) + q.rpr_key[rel]   -> mask -> softmax
// ctx    = a@v  (2x WMMA)  + a.rpr_value[rel]
// ---------------------------------------------------------------------------
__global__ void attention_kernel(const float* __restrict__ Q,
                                 const float* __restrict__ Km,
                                 const float* __restrict__ V,
                                 const float* __restrict__ rk,   // [17,32] layer slice
                                 const float* __restrict__ rv,   // [17,32] layer slice
                                 const int* __restrict__ lengths,
                                 float* __restrict__ Ctx) {
    __shared__ float smem[N_HEADS * 1792];   // per wave: q(512)+k(512)+v(512)+a(256)

    int n    = blockIdx.x;
    int tid  = threadIdx.x;
    int wave = tid >> 5;
    int lane = tid & 31;
    int head = wave;
    int half = lane >> 4;
    int lj   = lane & 15;
    int klane = lj + 16 * half;

    float* qs = smem + wave * 1792;
    float* ks = qs + 512;
    float* vs = ks + 512;
    float* as = vs + 512;

    int len = lengths[n];

    // load this head's q,k,v tiles [16 x 32]
#pragma unroll
    for (int i = 0; i < WLEN; ++i) {
        int g = (n * WLEN + i) * D_MODEL + head * DK + lane;
        qs[i * DK + lane] = Q[g];
        ks[i * DK + lane] = Km[g];
        vs[i * DK + lane] = V[g];
    }
    __syncthreads();

    // scores = q @ k^T  (single 16x16x32 WMMA)
    bf16x16 afrag, bfrag;
#pragma unroll
    for (int e = 0; e < 16; ++e) {
        afrag[e] = (__bf16)qs[lj * DK + kmapA(e, half)];
        bfrag[e] = (__bf16)ks[lj * DK + kmapA(e, half)];   // B[k=d][n=j] = k[j,d]
    }
    f32x8 sacc = {};
    sacc = __builtin_amdgcn_wmma_f32_16x16x32_bf16(
        false, afrag, false, bfrag, (short)0, sacc, false, false);

    // rpr-key term + key mask + softmax (rows live in 16-lane halves)
#pragma unroll
    for (int r = 0; r < 8; ++r) {
        int m = r + 8 * half;                 // query row
        float s = sacc[r];
        int rel = min(max(lj - m, -8), 8) + 8;
        const float* rkp = rk + rel * DK;
        float dot = 0.0f;
#pragma unroll
        for (int d = 0; d < DK; ++d) dot += qs[m * DK + d] * rkp[d];
        s += dot;
        if (lj >= len) s = NEGINF;
        float mx = s;
        for (int off = 1; off < 16; off <<= 1)
            mx = fmaxf(mx, __shfl_xor(mx, off, 16));
        float e = __expf(s - mx);
        float sum = e;
        for (int off = 1; off < 16; off <<= 1)
            sum += __shfl_xor(sum, off, 16);
        as[m * 16 + lj] = e / sum;
    }
    __syncthreads();

    // ctx = a @ v  (K padded 16->32 with zeros, two N-tiles of 16)
    bf16x16 afrag2;
#pragma unroll
    for (int e = 0; e < 16; ++e) {
        int k = kmapA(e, half);
        afrag2[e] = (k < 16) ? (__bf16)as[lj * 16 + k] : (__bf16)0.0f;
    }
    f32x8 cacc[2];
#pragma unroll
    for (int t = 0; t < 2; ++t) {
        bf16x16 bf;
#pragma unroll
        for (int e = 0; e < 16; ++e)
            bf[e] = (kmapA(e, half) < 16) ? (__bf16)vs[kmapA(e, half) * DK + t * 16 + lj]
                                 : (__bf16)0.0f;
        f32x8 z = {};
        cacc[t] = __builtin_amdgcn_wmma_f32_16x16x32_bf16(
            false, afrag2, false, bf, (short)0, z, false, false);
    }

    // rpr-value term + store ctx
#pragma unroll
    for (int t = 0; t < 2; ++t) {
#pragma unroll
        for (int r = 0; r < 8; ++r) {
            int m = r + 8 * half;
            int d = t * 16 + lj;
            float val = cacc[t][r];
#pragma unroll
            for (int j = 0; j < WLEN; ++j) {
                int rel = min(max(j - m, -8), 8) + 8;
                val += as[m * 16 + j] * rv[rel * DK + d];
            }
            Ctx[(n * WLEN + m) * D_MODEL + head * DK + d] = val;
        }
    }
}

// ---------------------------------------------------------------------------
// Host-side orchestration
// ---------------------------------------------------------------------------
extern "C" void kernel_launch(void* const* d_in, const int* in_sizes, int n_in,
                              void* d_out, int out_size, void* d_ws, size_t ws_size,
                              hipStream_t stream) {
    (void)in_sizes; (void)n_in; (void)out_size; (void)ws_size;

    const int*   xch     = (const int*)  d_in[0];
    const float* emb     = (const float*)d_in[1];
    const float* proj_w  = (const float*)d_in[2];
    const float* proj_b  = (const float*)d_in[3];
    const float* ln1_g   = (const float*)d_in[4];
    const float* ln1_b   = (const float*)d_in[5];
    const float* wq      = (const float*)d_in[6];
    const float* wq_b    = (const float*)d_in[7];
    const float* wk      = (const float*)d_in[8];
    const float* wk_b    = (const float*)d_in[9];
    const float* wv      = (const float*)d_in[10];
    const float* wv_b    = (const float*)d_in[11];
    const float* wo      = (const float*)d_in[12];
    const float* wo_b    = (const float*)d_in[13];
    const float* rpr_key = (const float*)d_in[14];
    const float* rpr_val = (const float*)d_in[15];
    const float* ln2_g   = (const float*)d_in[16];
    const float* ln2_b   = (const float*)d_in[17];
    const float* ff1_w   = (const float*)d_in[18];
    const float* ff1_b   = (const float*)d_in[19];
    const float* ff2_w   = (const float*)d_in[20];
    const float* ff2_b   = (const float*)d_in[21];
    const float* lnf_g   = (const float*)d_in[22];
    const float* lnf_b   = (const float*)d_in[23];
    float* out = (float*)d_out;

    // ---- workspace arena ----
    char* p = (char*)d_ws;
    float* x = (float*)p;  p += (size_t)MROWS * D_MODEL * sizeof(float);   // 32 MB
    float* h = (float*)p;  p += (size_t)MROWS * D_MODEL * sizeof(float);   // 32 MB
    float* big = (float*)p; p += (size_t)MROWS * DFF * sizeof(float);      // 128 MB (q/k/v or FF)
    float* q = big;
    float* k = big + (size_t)MROWS * D_MODEL;
    float* v = big + 2 * (size_t)MROWS * D_MODEL;
    __bf16* proj_bf = (__bf16*)p; p += 65536 * sizeof(__bf16);
    __bf16* wq_bf   = (__bf16*)p; p += NLAYERS * 65536 * sizeof(__bf16);
    __bf16* wk_bf   = (__bf16*)p; p += NLAYERS * 65536 * sizeof(__bf16);
    __bf16* wv_bf   = (__bf16*)p; p += NLAYERS * 65536 * sizeof(__bf16);
    __bf16* wo_bf   = (__bf16*)p; p += NLAYERS * 65536 * sizeof(__bf16);
    __bf16* ff1_bf  = (__bf16*)p; p += NLAYERS * 262144 * sizeof(__bf16);
    __bf16* ff2_bf  = (__bf16*)p; p += NLAYERS * 262144 * sizeof(__bf16);
    int* lengths    = (int*)p;    p += NWORDS * sizeof(int);

    // ---- one-time weight conversion f32 -> bf16 ----
    auto cvt = [&](const float* s, __bf16* d, int n) {
        cvt_bf16_kernel<<<(n + 255) / 256, 256, 0, stream>>>(s, d, n);
    };
    cvt(proj_w, proj_bf, 65536);
    cvt(wq,  wq_bf,  NLAYERS * 65536);
    cvt(wk,  wk_bf,  NLAYERS * 65536);
    cvt(wv,  wv_bf,  NLAYERS * 65536);
    cvt(wo,  wo_bf,  NLAYERS * 65536);
    cvt(ff1_w, ff1_bf, NLAYERS * 262144);
    cvt(ff2_w, ff2_bf, NLAYERS * 262144);

    lengths_kernel<<<(NWORDS + 255) / 256, 256, 0, stream>>>(xch, lengths, NWORDS);
    embed_kernel<<<MROWS, 256, 0, stream>>>(xch, emb, h);

    dim3 gridD(D_MODEL / BN, MROWS / BM);    // N=256  -> (4, 256)
    dim3 gridF(DFF / BN,     MROWS / BM);    // N=1024 -> (16, 256)

    // input projection: x = emb_rows @ proj + b
    gemm_bf16_wmma<0, 0><<<gridD, 256, 0, stream>>>(h, proj_bf, proj_b, nullptr, x,
                                                    MROWS, D_MODEL, D_MODEL);

    for (int l = 0; l < NLAYERS; ++l) {
        ln_kernel<<<MROWS, 256, 0, stream>>>(x, ln1_g + l * D_MODEL, ln1_b + l * D_MODEL, h);
        gemm_bf16_wmma<0, 0><<<gridD, 256, 0, stream>>>(h, wq_bf + l * 65536, wq_b + l * D_MODEL,
                                                        nullptr, q, MROWS, D_MODEL, D_MODEL);
        gemm_bf16_wmma<0, 0><<<gridD, 256, 0, stream>>>(h, wk_bf + l * 65536, wk_b + l * D_MODEL,
                                                        nullptr, k, MROWS, D_MODEL, D_MODEL);
        gemm_bf16_wmma<0, 0><<<gridD, 256, 0, stream>>>(h, wv_bf + l * 65536, wv_b + l * D_MODEL,
                                                        nullptr, v, MROWS, D_MODEL, D_MODEL);
        attention_kernel<<<NWORDS, 256, 0, stream>>>(q, k, v,
                                                     rpr_key + l * RPR_N * DK,
                                                     rpr_val + l * RPR_N * DK,
                                                     lengths, h);
        // x = x + ctx @ wo + b
        gemm_bf16_wmma<0, 1><<<gridD, 256, 0, stream>>>(h, wo_bf + l * 65536, wo_b + l * D_MODEL,
                                                        x, x, MROWS, D_MODEL, D_MODEL);
        ln_kernel<<<MROWS, 256, 0, stream>>>(x, ln2_g + l * D_MODEL, ln2_b + l * D_MODEL, h);
        // ff = relu(h @ ff1 + b1)
        gemm_bf16_wmma<1, 0><<<gridF, 256, 0, stream>>>(h, ff1_bf + l * 262144, ff1_b + l * DFF,
                                                        nullptr, big, MROWS, D_MODEL, DFF);
        // x = x + ff @ ff2 + b2
        gemm_bf16_wmma<0, 1><<<gridD, 256, 0, stream>>>(big, ff2_bf + l * 262144, ff2_b + l * D_MODEL,
                                                        x, x, MROWS, DFF, D_MODEL);
    }

    ln_kernel<<<MROWS, 256, 0, stream>>>(x, lnf_g, lnf_b, h);
    maxpool_kernel<<<NWORDS, 256, 0, stream>>>(h, out);
}
